// Keras_model_FE_RNN_17763984736692
// MI455X (gfx1250) — matrix operand order repaired and checked by reading, not compiled, not run on hardware
//
#include <hip/hip_runtime.h>
#include <hip/hip_bf16.h>

typedef __attribute__((ext_vector_type(16))) _Float16 v16h;
typedef __attribute__((ext_vector_type(8)))  _Float16 v8h;
typedef __attribute__((ext_vector_type(8)))  float    v8f;

#define BATCH 64
#define SEQT  512
#define FEAT  256
#define UNITS 128
#define NGATE 512                 // 4*UNITS
#define BT    (BATCH * SEQT)      // 32768

// Build a v16h WMMA fragment from two 16-byte chunks (LDS or global).
static __device__ __forceinline__ v16h ld_frag16(const _Float16* p0, const _Float16* p1) {
  v8h a = *(const v8h*)p0;
  v8h b = *(const v8h*)p1;
  v16h r;
#pragma unroll
  for (int i = 0; i < 8; ++i) { r[i] = a[i]; r[i + 8] = b[i]; }
  return r;
}

// ---------------- elementwise helpers ----------------

__global__ __launch_bounds__(256) void k_f32_to_f16(const float* __restrict__ s,
                                                    _Float16* __restrict__ d, int n) {
  for (int i = blockIdx.x * blockDim.x + threadIdx.x; i < n; i += gridDim.x * blockDim.x)
    d[i] = (_Float16)s[i];
}

// W [K,N] f32 row-major -> Wt [N,K] f16 (contiguous K per output column)
__global__ __launch_bounds__(256) void k_convT(const float* __restrict__ w,
                                               _Float16* __restrict__ wt, int K, int N) {
  int total = K * N;
  for (int i = blockIdx.x * blockDim.x + threadIdx.x; i < total; i += gridDim.x * blockDim.x) {
    int k = i / N, n = i - k * N;
    wt[(size_t)n * K + k] = (_Float16)w[i];
  }
}

// BatchNorm (inference) fused with f32->f16 conversion; C must be power of two (256)
__global__ __launch_bounds__(256) void k_bn16(const float* __restrict__ y,
                                              const float* __restrict__ g,
                                              const float* __restrict__ b,
                                              const float* __restrict__ m,
                                              const float* __restrict__ v,
                                              _Float16* __restrict__ d, int n, int C) {
  for (int i = blockIdx.x * blockDim.x + threadIdx.x; i < n; i += gridDim.x * blockDim.x) {
    int c = i & (C - 1);
    float sc = g[c] * rsqrtf(v[c] + 1e-3f);
    d[i] = (_Float16)((y[i] - m[c]) * sc + b[c]);
  }
}

// ---------------- bulk GEMM: Zx = X @ Wk ----------------
// X [BT, 256] f16 row-major, WkT [512, 256] f16 (K-contiguous). Output is written
// TIME-MAJOR: Zx[t][b][n] f16, so the recurrent kernel consumes one contiguous
// 64KB slab per step. Block tile 128x128, 8 waves (4M x 2N), double-buffered LDS,
// fully unrolled K loop (8 chunks of 32).
__global__ __launch_bounds__(256) void k_gemm_xwk(const _Float16* __restrict__ A,
                                                  const _Float16* __restrict__ Bt,
                                                  _Float16* __restrict__ C) {
  constexpr int K = FEAT;    // 256
  constexpr int N = NGATE;   // 512
  __shared__ _Float16 sA[2][128 * 32];
  __shared__ _Float16 sB[2][128 * 32];
  const int m0 = blockIdx.y * 128;
  const int n0 = blockIdx.x * 128;
  const int tid = threadIdx.x;
  const int lane = tid & 31;
  const int wid = tid >> 5;
  const int wm = wid >> 1;  // 0..3 -> 32-row strip
  const int wn = wid & 1;   // 0..1 -> 64-col strip
  const int ldr = tid >> 1;
  const int ldc = (tid & 1) * 16;

  const _Float16* ga = A + (size_t)(m0 + ldr) * K + ldc;
  const _Float16* gb = Bt + (size_t)(n0 + ldr) * K + ldc;

  // prime buffer 0
  v8h ra0 = *(const v8h*)ga;
  v8h ra1 = *(const v8h*)(ga + 8);
  v8h rb0 = *(const v8h*)gb;
  v8h rb1 = *(const v8h*)(gb + 8);
  *(v8h*)&sA[0][ldr * 32 + ldc]     = ra0;
  *(v8h*)&sA[0][ldr * 32 + ldc + 8] = ra1;
  *(v8h*)&sB[0][ldr * 32 + ldc]     = rb0;
  *(v8h*)&sB[0][ldr * 32 + ldc + 8] = rb1;

  v8f acc[2][4];
#pragma unroll
  for (int i = 0; i < 2; ++i)
#pragma unroll
    for (int j = 0; j < 4; ++j) acc[i][j] = (v8f){};

  const int arow = lane & 15;
  const int akb = (lane < 16) ? 0 : 8;
  const int bkb = (lane < 16) ? 0 : 8;

#pragma unroll
  for (int it = 0; it < K / 32; ++it) {
    __syncthreads();
    const int cur = it & 1;
    if (it + 1 < K / 32) {          // stage next chunk global -> VGPR (overlaps WMMA)
      const int kc = (it + 1) * 32;
      ra0 = *(const v8h*)(ga + kc);
      ra1 = *(const v8h*)(ga + kc + 8);
      rb0 = *(const v8h*)(gb + kc);
      rb1 = *(const v8h*)(gb + kc + 8);
    }
    v16h af[2], bf[4];
#pragma unroll
    for (int i = 0; i < 2; ++i) {
      const _Float16* p = &sA[cur][(wm * 32 + i * 16 + arow) * 32 + akb];
      af[i] = ld_frag16(p, p + 16);
    }
#pragma unroll
    for (int j = 0; j < 4; ++j) {
      const _Float16* p = &sB[cur][(wn * 64 + j * 16 + (lane & 15)) * 32 + bkb];
      bf[j] = ld_frag16(p, p + 16);
    }
#pragma unroll
    for (int i = 0; i < 2; ++i)
#pragma unroll
      for (int j = 0; j < 4; ++j)
        acc[i][j] = __builtin_amdgcn_wmma_f32_16x16x32_f16(
            false, af[i], false, bf[j], (short)0, acc[i][j], false, false);
    if (it + 1 < K / 32) {          // publish next chunk to the other buffer
      const int nxt = cur ^ 1;
      *(v8h*)&sA[nxt][ldr * 32 + ldc]     = ra0;
      *(v8h*)&sA[nxt][ldr * 32 + ldc + 8] = ra1;
      *(v8h*)&sB[nxt][ldr * 32 + ldc]     = rb0;
      *(v8h*)&sB[nxt][ldr * 32 + ldc + 8] = rb1;
    }
  }

  // epilogue: f32 -> f16, store time-major Zx[t][b][n]
  const int crow = (lane < 16) ? 0 : 8;
  const int ccol = lane & 15;
#pragma unroll
  for (int i = 0; i < 2; ++i)
#pragma unroll
    for (int j = 0; j < 4; ++j) {
      const int col = n0 + wn * 64 + j * 16 + ccol;
#pragma unroll
      for (int r = 0; r < 8; ++r) {
        const int m = m0 + wm * 32 + i * 16 + crow + r;   // m = b*SEQT + t
        const int b = m >> 9;
        const int t = m & (SEQT - 1);
        C[((size_t)t * BATCH + b) * N + col] = (_Float16)acc[i][j][r];
      }
    }
}

// ---------------- persistent recurrent LSTM (one 32-wave workgroup per direction) ----------------
// Wr fragments live in VGPRs (loop-invariant). LDS: h [64][128] f16 | gates [64][512] f32.
#define REC_LDS (64 * 128 * 2 + 64 * 512 * 4)   // 147456 B

__global__ __launch_bounds__(1024) void k_lstm_rec(const _Float16* __restrict__ Zxf,
                                                   const _Float16* __restrict__ Zxb,
                                                   const _Float16* __restrict__ WrTf,
                                                   const _Float16* __restrict__ WrTb,
                                                   const float* __restrict__ biasf,
                                                   const float* __restrict__ biasb,
                                                   float* __restrict__ Y,
                                                   int T, int ystride) {
  extern __shared__ char smem[];
  _Float16* sH = (_Float16*)smem;                       // 64*128 f16
  float*    sZ = (float*)(smem + 64 * 128 * 2);         // 64*512 f32

  const int dir = blockIdx.x;
  const _Float16* Zx  = dir ? Zxb : Zxf;    // time-major [T][64][512]
  const _Float16* WrT = dir ? WrTb : WrTf;  // [512][128]
  const float*    bia = dir ? biasb : biasf;
  const int dirOff = dir * UNITS;

  const int tid = threadIdx.x;
  const int lane = tid & 31;
  const int wid = tid >> 5;        // wave == 16-col n-tile of the 512 gate columns
  const int gate = wid >> 3;       // 0:i 1:f 2:g 3:o

  const int arow = lane & 15;
  const int akb = (lane < 16) ? 0 : 8;
  const int bkb = (lane < 16) ? 0 : 8;
  const int ccol = lane & 15;
  const int crow0 = (lane < 16) ? 0 : 8;

  // hoist this wave's Wr B-fragments into registers (loop-invariant)
  v16h bf[4];
#pragma unroll
  for (int kt = 0; kt < 4; ++kt) {
    const _Float16* bp = WrT + (size_t)(wid * 16 + (lane & 15)) * UNITS + kt * 32 + bkb;
    bf[kt] = ld_frag16(bp, bp + 16);
  }
  const float bv = bia[wid * 16 + ccol];

  {
    v8h z = (v8h){};
    *(v8h*)&sH[tid * 8] = z;       // 1024*8 == 64*128: full zero init of h
  }
  float c[8];
#pragma unroll
  for (int j = 0; j < 8; ++j) c[j] = 0.f;
  __syncthreads();

  for (int s = 0; s < T; ++s) {
    const int tt = dir ? (T - 1 - s) : s;

    // prefetch next step's contiguous 64KB Zx slab (64B per thread)
    if (s + 1 < T) {
      const int tn = dir ? (T - 2 - s) : (s + 1);
      __builtin_prefetch(Zx + (size_t)tn * BATCH * NGATE + tid * 32, 0, 1);
    }

    // ZR = h @ Wr for this wave's 16 columns (4 m-tiles, K=128 in 4 chunks)
    v8f acc[4];
#pragma unroll
    for (int mt = 0; mt < 4; ++mt) acc[mt] = (v8f){};
#pragma unroll
    for (int kt = 0; kt < 4; ++kt) {
#pragma unroll
      for (int mt = 0; mt < 4; ++mt) {
        const _Float16* ap = &sH[(mt * 16 + arow) * UNITS + kt * 32 + akb];
        v16h af = ld_frag16(ap, ap + 16);
        acc[mt] = __builtin_amdgcn_wmma_f32_16x16x32_f16(
            false, af, false, bf[kt], (short)0, acc[mt], false, false);
      }
    }

    // z = ZR + Zx + bias, per-gate activation, stash in LDS
    const _Float16* zrow = Zx + (size_t)tt * BATCH * NGATE + wid * 16 + ccol;
#pragma unroll
    for (int mt = 0; mt < 4; ++mt) {
#pragma unroll
      for (int r = 0; r < 8; ++r) {
        const int b = mt * 16 + crow0 + r;
        const float z = acc[mt][r] + (float)zrow[(size_t)b * NGATE] + bv;
        const float a = (gate == 2) ? fmaxf(z, 0.f) : (1.f / (1.f + __expf(-z)));
        sZ[b * NGATE + wid * 16 + ccol] = a;
      }
    }
    __syncthreads();

    // elementwise cell update: each thread owns 8 fixed (b,u) pairs
#pragma unroll
    for (int j = 0; j < 8; ++j) {
      const int q = tid + (j << 10);
      const int b = q >> 7;
      const int u = q & 127;
      const float iv = sZ[b * NGATE + u];
      const float fv = sZ[b * NGATE + 128 + u];
      const float gv = sZ[b * NGATE + 256 + u];
      const float ov = sZ[b * NGATE + 384 + u];
      const float cn = fv * c[j] + iv * gv;
      c[j] = cn;
      const float h = ov * fmaxf(cn, 0.f);
      sH[b * 128 + u] = (_Float16)h;
      Y[((size_t)b * T + tt) * ystride + dirOff + u] = h;
    }
    __syncthreads();
  }
}

// ---------------- final dense + sigmoid ----------------
__global__ __launch_bounds__(256) void k_dense(const float* __restrict__ Y,
                                               const float* __restrict__ Wd,
                                               const float* __restrict__ bd,
                                               float* __restrict__ out, int T) {
  __shared__ float red[256];
  const int b = blockIdx.x;
  const int c = threadIdx.x;
  const float h = (c < 128) ? Y[((size_t)b * T + (T - 1)) * 256 + c]
                            : Y[(size_t)b * T * 256 + c];
  red[c] = h * Wd[c];
  __syncthreads();
  for (int sft = 128; sft > 0; sft >>= 1) {
    if (c < sft) red[c] += red[c + sft];
    __syncthreads();
  }
  if (c == 0) out[b] = 1.f / (1.f + __expf(-(red[0] + bd[0])));
}

// ---------------- host orchestration ----------------
extern "C" void kernel_launch(void* const* d_in, const int* in_sizes, int n_in,
                              void* d_out, int out_size, void* d_ws, size_t ws_size,
                              hipStream_t stream) {
  (void)in_sizes; (void)n_in; (void)out_size; (void)ws_size;
  const float* x = (const float*)d_in[0];
  // params order: (l,dir) x {Wk,Wr,b} for l=0..2 dir=f,b ; bn0 g/b/m/v ; bn1 g/b/m/v ; dense_W ; dense_b
  const float* Wk[6];  const float* Wr[6];  const float* Bi[6];
  for (int i = 0; i < 6; ++i) {
    Wk[i] = (const float*)d_in[1 + 3 * i];
    Wr[i] = (const float*)d_in[2 + 3 * i];
    Bi[i] = (const float*)d_in[3 + 3 * i];
  }
  const float* bn[2][4];
  for (int l = 0; l < 2; ++l)
    for (int k = 0; k < 4; ++k) bn[l][k] = (const float*)d_in[19 + 4 * l + k];
  const float* dW = (const float*)d_in[27];
  const float* dB = (const float*)d_in[28];

  // workspace carve (256B aligned)
  char* w = (char*)d_ws;
  size_t off = 0;
  auto carve = [&](size_t bytes) { void* p = w + off; off = (off + bytes + 255) & ~(size_t)255; return p; };
  _Float16* x16  = (_Float16*)carve((size_t)BT * FEAT * 2);   // 16 MB
  _Float16* zxf  = (_Float16*)carve((size_t)BT * NGATE * 2);  // 32 MB (time-major)
  _Float16* zxb  = (_Float16*)carve((size_t)BT * NGATE * 2);  // 32 MB (time-major)
  float*    ybuf = (float*)carve((size_t)BT * 256 * 4);       // 32 MB
  _Float16* wkT[6]; _Float16* wrT[6];
  for (int i = 0; i < 6; ++i) {
    wkT[i] = (_Float16*)carve((size_t)FEAT * NGATE * 2);
    wrT[i] = (_Float16*)carve((size_t)UNITS * NGATE * 2);
  }

  hipFuncSetAttribute((const void*)k_lstm_rec,
                      hipFuncAttributeMaxDynamicSharedMemorySize, REC_LDS);

  const int NCONV = BT * FEAT;  // 8,388,608
  k_f32_to_f16<<<2048, 256, 0, stream>>>(x, x16, NCONV);
  for (int i = 0; i < 6; ++i) {
    k_convT<<<512, 256, 0, stream>>>(Wk[i], wkT[i], FEAT, NGATE);
    k_convT<<<256, 256, 0, stream>>>(Wr[i], wrT[i], UNITS, NGATE);
  }

  dim3 ggrid(NGATE / 128, BT / 128);  // (4, 256)
  for (int l = 0; l < 3; ++l) {
    k_gemm_xwk<<<ggrid, 256, 0, stream>>>(x16, wkT[2 * l],     zxf);
    k_gemm_xwk<<<ggrid, 256, 0, stream>>>(x16, wkT[2 * l + 1], zxb);
    k_lstm_rec<<<2, 1024, REC_LDS, stream>>>(zxf, zxb, wrT[2 * l], wrT[2 * l + 1],
                                             Bi[2 * l], Bi[2 * l + 1], ybuf, SEQT, 256);
    if (l < 2)
      k_bn16<<<2048, 256, 0, stream>>>(ybuf, bn[l][0], bn[l][1], bn[l][2], bn[l][3],
                                       x16, NCONV, 256);
  }
  k_dense<<<BATCH, 256, 0, stream>>>(ybuf, dW, dB, (float*)d_out, SEQT);
}
